// CNFAdapter_8512625181297
// MI455X (gfx1250) — compile-verified, matrix operand order checked by reading, not run on hardware
//
#include <hip/hip_runtime.h>
#include <cstdint>
#include <cstddef>

typedef _Float16 half_t;
typedef __attribute__((ext_vector_type(16))) _Float16 v16h;
typedef __attribute__((ext_vector_type(8)))  float    v8f;
typedef __attribute__((ext_vector_type(4)))  unsigned int u32x4;
typedef __attribute__((ext_vector_type(8)))  int      i32x8;
typedef __attribute__((ext_vector_type(4)))  int      i32x4;

#define WMMA_F32_F16(a, b, c) \
  __builtin_amdgcn_wmma_f32_16x16x32_f16(false, (a), false, (b), (short)0, (c), false, false)

#if defined(__has_builtin)
#if __has_builtin(__builtin_amdgcn_tensor_load_to_lds) && __has_builtin(__builtin_amdgcn_s_wait_tensorcnt)
#define HAVE_TDM 1
#endif
#endif

__device__ __forceinline__ float gelu_exact(float x) {
  return 0.5f * x * (1.0f + erff(x * 0.70710678118654752f));
}

// Escape the LDS base into an asm with a memory clobber: the compiler must
// assume the asm (standing in for the TDM engine) wrote the pointed-to LDS,
// so fragment loads stay alive and ordered after it. The loads themselves
// still address lsmem directly, keeping addrspace(3) provenance => ds_load.
__device__ __forceinline__ void lds_fence(const void* p) {
  asm volatile("" : : "v"(p) : "memory");
}

#if defined(HAVE_TDM)
// ---------------------------------------------------------------------------
// Tensor Data Mover: 2D tile (global -> LDS) descriptor per ISA 08_async_tensor
// D# group0: count=1(user), lds_addr, global_addr[56:0], type=2.
// D# group1: data_size=4B, pad (LDS-only) to rebuild padded row stride,
//            tensor_dim0/1 for OOB zero-fill, tile_dim0/1, dim0 stride.
// All units are dwords (data_size = 4B).
// ---------------------------------------------------------------------------
__device__ __forceinline__ void tdm_load_2d(unsigned lds_byte_addr, const void* gptr,
                                            unsigned tensor_d0, unsigned tensor_d1,
                                            unsigned tile_d0, unsigned tile_d1,
                                            unsigned stride_d0,
                                            unsigned pad_interval_log, unsigned pad_dwords)
{
  const unsigned long long ga = (unsigned long long)(uintptr_t)gptr;
  u32x4 g0;
  g0.x = 1u;                                         // count=1, user-mode descriptor
  g0.y = lds_byte_addr;                              // bits 63:32
  g0.z = (unsigned)(ga & 0xffffffffu);               // global_addr[31:0]
  g0.w = (unsigned)((ga >> 32) & 0x01ffffffu)        // global_addr[56:32]
         | (2u << 30);                               // type = 2 ("image")
  unsigned d0 = (2u << 16);                          // data_size = 4 bytes
  if (pad_dwords)
    d0 |= (1u << 20)                                 // pad_enable
        | (pad_interval_log << 22)                   // pad_interval (log2(dw)-1)
        | ((pad_dwords - 1u) << 25);                 // pad_amount (dwords-1)
  i32x8 g1;
  g1[0] = (int)d0;
  g1[1] = (int)((tensor_d0 & 0xffffu) << 16);        // [63:48] tensor_dim0 lo16
  g1[2] = (int)(((tensor_d0 >> 16) & 0xffffu)        // [79:64] tensor_dim0 hi16
         | ((tensor_d1 & 0xffffu) << 16));           // [95:80] tensor_dim1 lo16
  g1[3] = (int)(((tensor_d1 >> 16) & 0xffffu)        // [111:96] tensor_dim1 hi16
         | ((tile_d0 & 0xffffu) << 16));             // [127:112] tile_dim0
  g1[4] = (int)(tile_d1 & 0xffffu);                  // [143:128] tile_dim1 (tile_dim2=0)
  g1[5] = (int)stride_d0;                            // [191:160] dim0 stride lo32
  g1[6] = 0;                                         // stride hi16 | dim1 stride lo16
  g1[7] = 0;                                         // dim1 stride hi32
  const i32x4 z4 = {0, 0, 0, 0};
#if defined(__clang_major__) && (__clang_major__ >= 23)
  const i32x8 z8 = {0, 0, 0, 0, 0, 0, 0, 0};
  __builtin_amdgcn_tensor_load_to_lds(g0, g1, z4, z4, z8, 0);
#else
  __builtin_amdgcn_tensor_load_to_lds(g0, g1, z4, z4, 0);
#endif
}
#endif  // HAVE_TDM

// ---------------------------------------------------------------------------
// Embedding gather + masked mean pool:  x[b,c,:] -> f32 and f16 copies
// ---------------------------------------------------------------------------
__global__ __launch_bounds__(256)
void k_embed_pool(const int* __restrict__ var_idx, const int* __restrict__ sign_idx,
                  const unsigned char* __restrict__ lit_mask,
                  const float* __restrict__ var_emb, const float* __restrict__ sign_emb,
                  const float* __restrict__ litpos_emb, const float* __restrict__ clausepos_emb,
                  float* __restrict__ xf32, half_t* __restrict__ xf16,
                  int C, int L, int D)
{
  const int row = blockIdx.x;         // b*C + c
  const int c = row % C;
  const int d = threadIdx.x;
  const int base = row * L;
  float acc = 0.f;
  int cnt = 0;
  for (int l = 0; l < L; ++l) {
    if (lit_mask[base + l]) {
      const int v = var_idx[base + l];
      const int s = sign_idx[base + l];
      acc += var_emb[v * D + d] + sign_emb[s * D + d] + litpos_emb[l * D + d];
      ++cnt;
    }
  }
  const float x = acc / (float)cnt + clausepos_emb[c * D + d];
  xf32[(size_t)row * D + d] = x;
  xf16[(size_t)row * D + d] = (half_t)x;
}

// ---------------------------------------------------------------------------
// f32 -> f16 converter (weights, pq)
// ---------------------------------------------------------------------------
__global__ void k_f32_to_f16(const float* __restrict__ src, half_t* __restrict__ dst, size_t n) {
  size_t i = (size_t)blockIdx.x * blockDim.x + threadIdx.x;
  const size_t stride = (size_t)gridDim.x * blockDim.x;
  for (; i < n; i += stride) dst[i] = (half_t)src[i];
}

// ---------------------------------------------------------------------------
// WMMA GEMM:  C[M,N] = A[M,K](f16) * W[N,K]^T(f16) + bias, optional exact GELU.
// Tile: BM=128 x BN=64 x BK=64, 128 threads (4 waves), each wave owns 32x64.
// Tiles arrive via TDM (double-buffered, wave0-issued, TENSORcnt-tracked);
// TDM pad_enable rebuilds the 80-half padded LDS row stride and OOB rows
// zero-fill in hardware. Fallback: cooperative vector loads.
// ---------------------------------------------------------------------------
#define BM 128
#define BN 64
#define BK 64
#define LSTR 80                      // halves per LDS row (64 + 16 pad)
#define LDS_A_OFF(buf) ((unsigned)(buf) * (BM * LSTR))
#define LDS_B_OFF(buf) ((unsigned)(2 * BM * LSTR + (buf) * (BN * LSTR)))

__global__ __launch_bounds__(128)
void k_gemm(const half_t* __restrict__ A, int M, int K,
            const half_t* __restrict__ W, int N,
            const float* __restrict__ bias,
            float* __restrict__ Cf32, half_t* __restrict__ Cf16,
            int act)
{
  // Single shared block => LDS offsets below are exact byte addresses for D#.
  __shared__ __align__(128) half_t lsmem[2 * (BM + BN) * LSTR];

  const int tid = threadIdx.x;
  const int lane = tid & 31;
  const int wave = tid >> 5;
  const int l15 = lane & 15;
  const int khalf = lane >> 4;
  const int m0 = blockIdx.y * BM;
  const int n0 = blockIdx.x * BN;
  const int ksteps = K / BK;

  v8f zero = {};
  v8f cacc[2][4];
#pragma unroll
  for (int s = 0; s < 2; ++s)
#pragma unroll
    for (int n = 0; n < 4; ++n) cacc[s][n] = zero;

#if defined(HAVE_TDM)
  // issue tile kt into buffer (kt & 1): A 128x(64h=32dw) rows, B 64x32dw rows
  auto issue = [&](int kt) {
    const int buf = kt & 1;
    const int k0 = kt * BK;
    tdm_load_2d(LDS_A_OFF(buf) * 2u, A + (size_t)m0 * K + k0,
                (unsigned)(K >> 1), (unsigned)(M - m0),
                BK / 2, BM, (unsigned)(K >> 1), /*pad 32dw*/ 4, /*pad*/ 8);
    tdm_load_2d(LDS_B_OFF(buf) * 2u, W + (size_t)n0 * K + k0,
                (unsigned)(K >> 1), (unsigned)(N - n0),
                BK / 2, BN, (unsigned)(K >> 1), 4, 8);
  };
  if (wave == 0) issue(0);
#endif

  for (int kt = 0; kt < ksteps; ++kt) {
    const int cur = kt & 1;
#if defined(HAVE_TDM)
    if (wave == 0) {
      if (kt + 1 < ksteps) {
        issue(kt + 1);                               // overlap next tile's DMA
        __builtin_amdgcn_s_wait_tensorcnt(2);        // current tile's 2 ops done
      } else {
        __builtin_amdgcn_s_wait_tensorcnt(0);
      }
    }
    __syncthreads();
#else
    // Fallback: cooperative load of tile kt into buffer cur.
    __syncthreads();
    {
      const int k0 = kt * BK;
      half_t* la = &lsmem[LDS_A_OFF(cur)];
      half_t* lb = &lsmem[LDS_B_OFF(cur)];
#pragma unroll
      for (int it = 0; it < 8; ++it) {               // A: 128x64 halves
        const int idx = (tid + it * 128) * 8;
        const int r = idx >> 6, cc = idx & 63;
        uint4 va = {0u, 0u, 0u, 0u};
        if (m0 + r < M) va = *reinterpret_cast<const uint4*>(A + (size_t)(m0 + r) * K + k0 + cc);
        *reinterpret_cast<uint4*>(&la[r * LSTR + cc]) = va;
      }
#pragma unroll
      for (int it = 0; it < 4; ++it) {               // B: 64x64 halves
        const int idx = (tid + it * 128) * 8;
        const int r = idx >> 6, cc = idx & 63;
        uint4 vb = {0u, 0u, 0u, 0u};
        if (n0 + r < N) vb = *reinterpret_cast<const uint4*>(W + (size_t)(n0 + r) * K + k0 + cc);
        *reinterpret_cast<uint4*>(&lb[r * LSTR + cc]) = vb;
      }
    }
    __syncthreads();
#endif
    // Compiler-level fence: lsmem escapes here with a memory clobber, so the
    // fragment loads below cannot fold/hoist, yet keep LDS provenance (ds ops).
    lds_fence(lsmem);
    const unsigned aoff = LDS_A_OFF(cur);
    const unsigned boff = LDS_B_OFF(cur);
#pragma unroll
    for (int ks = 0; ks < 2; ++ks) {                 // two 32-k WMMA steps per tile
      const v16h af0 = *reinterpret_cast<const v16h*>(
          &lsmem[aoff + (wave * 32 + l15) * LSTR + ks * 32 + khalf * 16]);
      const v16h af1 = *reinterpret_cast<const v16h*>(
          &lsmem[aoff + (wave * 32 + 16 + l15) * LSTR + ks * 32 + khalf * 16]);
#pragma unroll
      for (int n = 0; n < 4; ++n) {
        const v16h bf = *reinterpret_cast<const v16h*>(
            &lsmem[boff + (n * 16 + l15) * LSTR + ks * 32 + khalf * 16]);
        cacc[0][n] = WMMA_F32_F16(af0, bf, cacc[0][n]);
        cacc[1][n] = WMMA_F32_F16(af1, bf, cacc[1][n]);
      }
    }
    __syncthreads();
  }

  // Epilogue (C layout: vgpr r -> row r / r+8 by lane half; col = lane&15)
#pragma unroll
  for (int s = 0; s < 2; ++s) {
    const int rowb = m0 + wave * 32 + s * 16 + khalf * 8;
#pragma unroll
    for (int n = 0; n < 4; ++n) {
      const int col = n0 + n * 16 + l15;
      const float bb = (col < N) ? bias[col] : 0.f;
#pragma unroll
      for (int r = 0; r < 8; ++r) {
        const int row = rowb + r;
        float vv = cacc[s][n][r] + bb;
        if (act == 1) vv = gelu_exact(vv);
        if (row < M && col < N) {
          const size_t o = (size_t)row * N + col;
          if (Cf32) Cf32[o] = vv;
          if (Cf16) Cf16[o] = (half_t)vv;
        }
      }
    }
  }
}

// ---------------------------------------------------------------------------
// Flash attention, one wave per (q-tile of 16, head, batch).
// DH = head dim (32 encoder, 128 prefix). 32-key chunks:
//   S = Q*K^T via chained WMMA (k-dim = DH), online softmax in C-layout,
//   P staged via LDS to become a WMMA A-operand, O += P*V via WMMA.
// ---------------------------------------------------------------------------
template <int DH>
__global__ __launch_bounds__(32)
void attn_flash(const half_t* __restrict__ Q, const half_t* __restrict__ K,
                const half_t* __restrict__ V, half_t* __restrict__ O,
                int ldq, int ldkv, int ldo, int nkv, int qlen, int qPerBatch,
                float scale)
{
  constexpr int KH = DH / 32;   // k-steps per score tile
  constexpr int NJ = DH / 16;   // output column tiles
  __shared__ __align__(32) half_t lp[16 * 32];

  const int qt = blockIdx.x, h = blockIdx.y, b = blockIdx.z;
  const int lane = threadIdx.x;
  const int l15 = lane & 15;
  const int khalf = lane >> 4;

  const half_t* Qb = Q + (size_t)((qPerBatch ? b * qlen : 0) + qt * 16) * ldq + h * DH;
  const half_t* Kb = K + (size_t)b * nkv * ldkv + h * DH;
  const half_t* Vb = V + (size_t)b * nkv * ldkv + h * DH;
  half_t*       Ob = O + (size_t)(b * qlen + qt * 16) * ldo + h * DH;

  v16h qf[KH];
#pragma unroll
  for (int i = 0; i < KH; ++i)
    qf[i] = *reinterpret_cast<const v16h*>(Qb + (size_t)l15 * ldq + i * 32 + khalf * 16);

  float mrow[8], lrow[8];
  v8f zero = {};
  v8f acc[NJ];
#pragma unroll
  for (int r = 0; r < 8; ++r) { mrow[r] = -1e30f; lrow[r] = 0.f; }
#pragma unroll
  for (int j = 0; j < NJ; ++j) acc[j] = zero;

  for (int k0 = 0; k0 < nkv; k0 += 32) {
    // ---- scores for 32 keys (two 16-key WMMA tiles) ----
    v8f s0 = {}, s1 = {};
#pragma unroll
    for (int i = 0; i < KH; ++i) {
      const v16h kf0 = *reinterpret_cast<const v16h*>(
          Kb + (size_t)(k0 + l15) * ldkv + i * 32 + khalf * 16);
      const v16h kf1 = *reinterpret_cast<const v16h*>(
          Kb + (size_t)(k0 + 16 + l15) * ldkv + i * 32 + khalf * 16);
      s0 = WMMA_F32_F16(qf[i], kf0, s0);
      s1 = WMMA_F32_F16(qf[i], kf1, s1);
    }
    // ---- online softmax across the 32 new columns ----
#pragma unroll
    for (int r = 0; r < 8; ++r) {
      const float a0 = s0[r] * scale;
      const float a1 = s1[r] * scale;
      float tm = fmaxf(a0, a1);
#pragma unroll
      for (int o = 8; o > 0; o >>= 1) tm = fmaxf(tm, __shfl_xor(tm, o, 16));
      const float mn = fmaxf(mrow[r], tm);
      const float p0 = __expf(a0 - mn);
      const float p1 = __expf(a1 - mn);
      float ls = p0 + p1;
#pragma unroll
      for (int o = 8; o > 0; o >>= 1) ls += __shfl_xor(ls, o, 16);
      const float f = __expf(mrow[r] - mn);
      lrow[r] = lrow[r] * f + ls;
      mrow[r] = mn;
#pragma unroll
      for (int j = 0; j < NJ; ++j) acc[j][r] = acc[j][r] * f;
      const int prow = r + khalf * 8;
      lp[prow * 32 + l15] = (half_t)p0;
      lp[prow * 32 + 16 + l15] = (half_t)p1;
    }
    __syncthreads();
    // P as A-operand: 16 rows x 32 keys
    const v16h pf = *reinterpret_cast<const v16h*>(&lp[l15 * 32 + khalf * 16]);
#pragma unroll
    for (int j = 0; j < NJ; ++j) {
      union { v16h v; half_t e[16]; } bv;
#pragma unroll
      for (int kk = 0; kk < 16; ++kk)
        bv.e[kk] = Vb[(size_t)(k0 + khalf * 16 + kk) * ldkv + j * 16 + l15];
      acc[j] = WMMA_F32_F16(pf, bv.v, acc[j]);
    }
    __syncthreads();
  }
  // ---- normalize and store (f16, feeds out-proj GEMM) ----
#pragma unroll
  for (int j = 0; j < NJ; ++j) {
#pragma unroll
    for (int r = 0; r < 8; ++r) {
      const float outv = acc[j][r] / lrow[r];
      Ob[(size_t)(r + khalf * 8) * ldo + j * 16 + l15] = (half_t)outv;
    }
  }
}

// ---------------------------------------------------------------------------
// LayerNorm (optionally residual add). 128 threads = 4 waves, one wave per row.
// ---------------------------------------------------------------------------
__global__ __launch_bounds__(128)
void k_ln(const float* __restrict__ A, const float* __restrict__ Bsrc, int bMod,
          const float* __restrict__ w, const float* __restrict__ bias,
          float* __restrict__ o32, half_t* __restrict__ o16, int dim)
{
  const int row = blockIdx.x * 4 + (threadIdx.x >> 5);
  const int lane = threadIdx.x & 31;
  const int n = dim >> 5;
  float v[32];
  const float* ar = A + (size_t)row * dim;
  const float* br = Bsrc ? (Bsrc + (size_t)(bMod ? (row % bMod) : row) * dim) : nullptr;
  float s = 0.f;
  for (int i = 0; i < n; ++i) {
    float x = ar[i * 32 + lane];
    if (br) x += br[i * 32 + lane];
    v[i] = x;
    s += x;
  }
#pragma unroll
  for (int o = 16; o > 0; o >>= 1) s += __shfl_xor(s, o, 32);
  const float mean = s / (float)dim;
  float vs = 0.f;
  for (int i = 0; i < n; ++i) { const float d = v[i] - mean; vs += d * d; }
#pragma unroll
  for (int o = 16; o > 0; o >>= 1) vs += __shfl_xor(vs, o, 32);
  const float rstd = rsqrtf(vs / (float)dim + 1e-5f);
  for (int i = 0; i < n; ++i) {
    const int c = i * 32 + lane;
    const float y = (v[i] - mean) * rstd * w[c] + bias[c];
    if (o32) o32[(size_t)row * dim + c] = y;
    if (o16) o16[(size_t)row * dim + c] = (half_t)y;
  }
}

// ---------------------------------------------------------------------------
// Orchestration
// ---------------------------------------------------------------------------
extern "C" void kernel_launch(void* const* d_in, const int* in_sizes, int n_in,
                              void* d_out, int out_size, void* d_ws, size_t ws_size,
                              hipStream_t stream)
{
  (void)in_sizes; (void)n_in; (void)out_size; (void)ws_size;
  const int*   var_idx  = (const int*)d_in[0];
  const int*   sign_idx = (const int*)d_in[1];
  const unsigned char* lit_mask = (const unsigned char*)d_in[2];
  const float* var_emb  = (const float*)d_in[3];
  const float* sign_emb = (const float*)d_in[4];
  const float* litpos   = (const float*)d_in[5];
  const float* clpos    = (const float*)d_in[6];
  const float* enc_in_w  = (const float*)d_in[7];
  const float* enc_in_b  = (const float*)d_in[8];
  const float* enc_out_w = (const float*)d_in[9];
  const float* enc_out_b = (const float*)d_in[10];
  const float* ff1_w = (const float*)d_in[11];
  const float* ff1_b = (const float*)d_in[12];
  const float* ff2_w = (const float*)d_in[13];
  const float* ff2_b = (const float*)d_in[14];
  const float* n1_w = (const float*)d_in[15];
  const float* n1_b = (const float*)d_in[16];
  const float* n2_w = (const float*)d_in[17];
  const float* n2_b = (const float*)d_in[18];
  const float* th_ln_w = (const float*)d_in[19];
  const float* th_ln_b = (const float*)d_in[20];
  const float* th_w = (const float*)d_in[21];
  const float* th_b = (const float*)d_in[22];
  const float* pq   = (const float*)d_in[23];
  const float* pa_in_w  = (const float*)d_in[24];
  const float* pa_in_b  = (const float*)d_in[25];
  const float* pa_out_w = (const float*)d_in[26];
  const float* pa_out_b = (const float*)d_in[27];
  const float* pn_w = (const float*)d_in[28];
  const float* pn_b = (const float*)d_in[29];

  constexpr int B = 32, Cc = 512, L = 16, D = 256, H = 1024, P = 16, NHh = 8, NLl = 4;
  const int M = B * Cc;  // 16384 clause rows

  // ---- workspace carve (256B aligned regions) ----
  char* wbase = (char*)d_ws;
  size_t off = 0;
  auto alloc = [&](size_t bytes) -> char* {
    char* p = wbase + off;
    off = (off + bytes + 255) & ~(size_t)255;
    return p;
  };
  half_t* encInW16  = (half_t*)alloc((size_t)NLl * 3 * D * D * 2);
  half_t* encOutW16 = (half_t*)alloc((size_t)NLl * D * D * 2);
  half_t* ff1W16    = (half_t*)alloc((size_t)NLl * 4 * D * D * 2);
  half_t* ff2W16    = (half_t*)alloc((size_t)NLl * 4 * D * D * 2);
  half_t* thW16     = (half_t*)alloc((size_t)H * D * 2);
  half_t* paInW16   = (half_t*)alloc((size_t)3 * H * H * 2);
  half_t* paOutW16  = (half_t*)alloc((size_t)H * H * 2);
  half_t* pqf16     = (half_t*)alloc((size_t)P * H * 2);
  float*  xf32   = (float*) alloc((size_t)M * D * 4);
  half_t* xf16   = (half_t*)alloc((size_t)M * D * 2);
  half_t* qf16   = (half_t*)alloc((size_t)M * D * 2);
  half_t* kf16   = (half_t*)alloc((size_t)M * D * 2);
  half_t* vf16   = (half_t*)alloc((size_t)M * D * 2);
  half_t* aOf16  = (half_t*)alloc((size_t)M * D * 2);
  float*  tmpf32 = (float*) alloc((size_t)M * D * 4);
  half_t* ff1f16 = (half_t*)alloc((size_t)M * 4 * D * 2);
  half_t* chf16  = (half_t*)alloc((size_t)M * H * 2);
  half_t* kpf16  = (half_t*)alloc((size_t)M * H * 2);
  half_t* vpf16  = (half_t*)alloc((size_t)M * H * 2);
  half_t* q0f16  = (half_t*)alloc((size_t)P * H * 2);
  half_t* pAf16  = (half_t*)alloc((size_t)B * P * H * 2);
  float*  pOf32  = (float*) alloc((size_t)B * P * H * 4);

  auto cvt = [&](const float* s, half_t* dst, size_t n) {
    int blocks = (int)((n + 255) / 256);
    if (blocks > 2048) blocks = 2048;
    k_f32_to_f16<<<blocks, 256, 0, stream>>>(s, dst, n);
  };
  cvt(enc_in_w,  encInW16,  (size_t)NLl * 3 * D * D);
  cvt(enc_out_w, encOutW16, (size_t)NLl * D * D);
  cvt(ff1_w,     ff1W16,    (size_t)NLl * 4 * D * D);
  cvt(ff2_w,     ff2W16,    (size_t)NLl * 4 * D * D);
  cvt(th_w,      thW16,     (size_t)H * D);
  cvt(pa_in_w,   paInW16,   (size_t)3 * H * H);
  cvt(pa_out_w,  paOutW16,  (size_t)H * H);
  cvt(pq,        pqf16,     (size_t)P * H);

  auto gemm = [&](const half_t* A, int Mm, int Kk, const half_t* W, int Nn,
                  const float* bias, float* c32, half_t* c16, int act) {
    dim3 g((Nn + BN - 1) / BN, (Mm + BM - 1) / BM);
    k_gemm<<<g, 128, 0, stream>>>(A, Mm, Kk, W, Nn, bias, c32, c16, act);
  };

  // ---- embedding + masked mean pool ----
  k_embed_pool<<<M, D, 0, stream>>>(var_idx, sign_idx, lit_mask, var_emb, sign_emb,
                                    litpos, clpos, xf32, xf16, Cc, L, D);

  // ---- encoder layers (post-norm, exact GELU) ----
  for (int i = 0; i < NLl; ++i) {
    const half_t* Wq = encInW16 + (size_t)i * 3 * D * D;
    const half_t* Wk = Wq + (size_t)D * D;
    const half_t* Wv = Wk + (size_t)D * D;
    const float* bq = enc_in_b + (size_t)i * 3 * D;
    gemm(xf16, M, D, Wq, D, bq,         nullptr, qf16, 0);
    gemm(xf16, M, D, Wk, D, bq + D,     nullptr, kf16, 0);
    gemm(xf16, M, D, Wv, D, bq + 2 * D, nullptr, vf16, 0);
    attn_flash<32><<<dim3(Cc / 16, NHh, B), 32, 0, stream>>>(
        qf16, kf16, vf16, aOf16, D, D, D, Cc, Cc, 1, 0.17677669529663687f);
    gemm(aOf16, M, D, encOutW16 + (size_t)i * D * D, D,
         enc_out_b + (size_t)i * D, tmpf32, nullptr, 0);
    k_ln<<<M / 4, 128, 0, stream>>>(xf32, tmpf32, 0, n1_w + (size_t)i * D, n1_b + (size_t)i * D,
                                    xf32, xf16, D);
    gemm(xf16, M, D, ff1W16 + (size_t)i * 4 * D * D, 4 * D,
         ff1_b + (size_t)i * 4 * D, nullptr, ff1f16, 1);
    gemm(ff1f16, M, 4 * D, ff2W16 + (size_t)i * 4 * D * D, D,
         ff2_b + (size_t)i * D, tmpf32, nullptr, 0);
    k_ln<<<M / 4, 128, 0, stream>>>(xf32, tmpf32, 0, n2_w + (size_t)i * D, n2_b + (size_t)i * D,
                                    xf32, xf16, D);
  }

  // ---- to_hidden: ch = LN(x) @ th_w^T + th_b  (f16 for downstream GEMMs) ----
  k_ln<<<M / 4, 128, 0, stream>>>(xf32, nullptr, 0, th_ln_w, th_ln_b, nullptr, xf16, D);
  gemm(xf16, M, D, thW16, H, th_b, nullptr, chf16, 0);

  // ---- prefix cross-attention ----
  gemm(pqf16, P, H, paInW16, H, pa_in_b, nullptr, q0f16, 0);                           // Wq
  gemm(chf16, M, H, paInW16 + (size_t)H * H, H, pa_in_b + H, nullptr, kpf16, 0);       // Wk
  gemm(chf16, M, H, paInW16 + (size_t)2 * H * H, H, pa_in_b + 2 * H, nullptr, vpf16, 0); // Wv
  attn_flash<128><<<dim3(1, NHh, B), 32, 0, stream>>>(
      q0f16, kpf16, vpf16, pAf16, H, H, H, Cc, P, 0, 0.08838834764831845f);
  gemm(pAf16, B * P, H, paOutW16, H, pa_out_b, pOf32, nullptr, 0);

  // ---- final residual + LN -> d_out (f32) ----
  k_ln<<<B * P / 4, 128, 0, stream>>>(pOf32, pq, P, pn_w, pn_b, (float*)d_out, nullptr, H);
}
